// Atten_Head_29145648071315
// MI455X (gfx1250) — matrix operand order repaired and checked by reading, not compiled, not run on hardware
//
#include <hip/hip_runtime.h>
#include <stdint.h>

typedef __attribute__((ext_vector_type(16))) __bf16 v16bf;
typedef __attribute__((ext_vector_type(8)))  __bf16 v8bf;
typedef __attribute__((ext_vector_type(2)))  __bf16 v2bf;
typedef __attribute__((ext_vector_type(8)))  float  v8f;
typedef __attribute__((ext_vector_type(4)))  float  v4f;

// ---- fp32 -> packed 2x bf16. Try HW paths first (builtin, then fptrunc). ---
__device__ __forceinline__ uint32_t cvt_pk2(float a, float b) {
#if __has_builtin(__builtin_amdgcn_cvt_pk_bf16_f32)
  v2bf p = __builtin_amdgcn_cvt_pk_bf16_f32(a, b);
  return __builtin_bit_cast(uint32_t, p);
#else
  v2bf p = { (__bf16)a, (__bf16)b };   // fptrunc: backend picks native cvt
  return __builtin_bit_cast(uint32_t, p);
#endif
}
__device__ __forceinline__ __bf16 f2bf(float f) { return (__bf16)f; }

// 16 floats (two v8f) -> one v16bf A-operand
__device__ __forceinline__ v16bf cvt16(v8f f0, v8f f1) {
  union { uint32_t u[8]; v16bf v; } c;
#pragma unroll
  for (int i = 0; i < 4; ++i) {
    c.u[i]     = cvt_pk2(f0[2 * i], f0[2 * i + 1]);
    c.u[i + 4] = cvt_pk2(f1[2 * i], f1[2 * i + 1]);
  }
  return c.v;
}

// ---------------- Kernel 0: weight [1024,64] fp32 -> W^T [64,1024] bf16 -------
__global__ void wt_prep(const float* __restrict__ W, __bf16* __restrict__ Wt) {
  int idx = blockIdx.x * blockDim.x + threadIdx.x;   // 65536 elements
  int d = idx >> 6, h = idx & 63;
  Wt[h * 1024 + d] = f2bf(W[d * 64 + h]);
}

// ---------------- Kernel 1: projection  Y[B*S,64] = X[B*S,1024] @ W ----------
// One block = 128 rows (8 waves x 16-row M-tile). K loop 1024 in steps of 32.
// A operand: lane M=l&15, half=l>>4; elems 0..7 = K[16h..16h+8), 8..15 = K[16h+8..16h+16)
// B operand: lane N=l&15; elems 0..15 = K[16*half .. +16) -> contiguous W^T row.
__global__ __launch_bounds__(256) void proj_kernel(
    const float* __restrict__ X, const __bf16* __restrict__ Wt,
    __bf16* __restrict__ Yrow,     // row-major [B*S][64]  (Q, K) or nullptr
    __bf16* __restrict__ Ycol)     // transposed [B][64][2048] (V)  or nullptr
{
  const int lane = threadIdx.x & 31;
  const int wv   = threadIdx.x >> 5;
  const int n    = lane & 15;
  const int half = lane >> 4;
  const int row0 = blockIdx.x * 128 + wv * 16;
  const float* arow  = X + (size_t)(row0 + n) * 1024;
  const __bf16* wrow = Wt + (size_t)n * 1024 + 16 * half;   // + nt*16*1024 + kb

  v8f acc[4] = {};
  for (int kb = 0; kb < 1024; kb += 32) {
    __builtin_prefetch(arow + kb + 128, 0, 0);       // ~4 iterations ahead
    // issue all loads first -> one clause, one wait
    v8f f0 = *(const v8f*)(arow + kb + 16 * half);
    v8f f1 = *(const v8f*)(arow + kb + 16 * half + 8);
    v16bf bm[4];
#pragma unroll
    for (int nt = 0; nt < 4; ++nt)
      bm[nt] = *(const v16bf*)(wrow + (size_t)nt * 16 * 1024 + kb);
    v16bf a = cvt16(f0, f1);
#pragma unroll
    for (int nt = 0; nt < 4; ++nt)
      acc[nt] = __builtin_amdgcn_wmma_f32_16x16x32_bf16(
          false, a, false, bm[nt], (short)0, acc[nt], false, false);
  }
  // C/D layout: lane holds column h = nt*16+n, rows M = 8*half + r
  if (Yrow) {
#pragma unroll
    for (int nt = 0; nt < 4; ++nt) {
      int h = nt * 16 + n;
#pragma unroll
      for (int r = 0; r < 8; ++r)
        Yrow[(size_t)(row0 + 8 * half + r) * 64 + h] = f2bf(acc[nt][r]);
    }
  } else {
    int b = row0 >> 11;                       // 2048 rows per batch
    int sbase = (row0 & 2047) + 8 * half;     // even -> u32-aligned packed store
    __bf16* base = Ycol + (size_t)b * 64 * 2048;
#pragma unroll
    for (int nt = 0; nt < 4; ++nt) {
      int h = nt * 16 + n;
      uint32_t* dst = (uint32_t*)(base + (size_t)h * 2048 + sbase);
#pragma unroll
      for (int r = 0; r < 8; r += 2)
        dst[r >> 1] = cvt_pk2(acc[nt][r], acc[nt][r + 1]);
    }
  }
}

// ---------------- Kernel 2: causal flash attention ---------------------------
// Per wave: 16 queries. Computes S^T = K·Q^T (so softmax rows reduce with one
// shfl_xor(16)), online softmax, P^T staged to LDS as bf16, O^T += V^T·P^T.
__global__ __launch_bounds__(256) void attn_kernel(
    const __bf16* __restrict__ Qb, const __bf16* __restrict__ Kb,
    const __bf16* __restrict__ Vt, float* __restrict__ Out)
{
  __shared__ __align__(32) uint8_t smem[8 * 1024];   // 16q x 32k bf16 per wave
  const int lane = threadIdx.x & 31;
  const int wv   = threadIdx.x >> 5;
  const int n    = lane & 15;
  const int half = lane >> 4;
  const int b    = blockIdx.y;
  const int q0   = blockIdx.x * 128 + wv * 16;
  const __bf16* QB = Qb + (size_t)b * 2048 * 64;
  const __bf16* KB = Kb + (size_t)b * 2048 * 64;
  const __bf16* VT = Vt + (size_t)b * 64 * 2048;
  __bf16* lds = (__bf16*)(smem + wv * 1024);

  // Q^T B-operand, reused every chunk: lane column q=q0+n, K=h contiguous.
  v16bf qB[2];
#pragma unroll
  for (int hc = 0; hc < 2; ++hc)
    qB[hc] = *(const v16bf*)(QB + (size_t)(q0 + n) * 64 + hc * 32 + 16 * half);

  v8f o[4] = {};
  float m = -__builtin_inff(), lsum = 0.f;
  const float SC_LOG2E = 0.0220970869120796f * 1.44269504088896340736f; // 1/sqrt(2048)*log2e
  const int qq   = q0 + n;
  const int kend = q0 + 16;

  for (int kb = 0; kb < kend; kb += 32) {
    // ---- S^T tiles: keys [kb,kb+16) and [kb+16,kb+32) vs 16 queries --------
    v8f st[2] = {};
#pragma unroll
    for (int t = 0; t < 2; ++t) {
      const __bf16* krow = KB + (size_t)(kb + t * 16 + n) * 64;
#pragma unroll
      for (int hc = 0; hc < 2; ++hc) {
        v8bf lo = *(const v8bf*)(krow + hc * 32 + 16 * half);
        v8bf hi = *(const v8bf*)(krow + hc * 32 + 16 * half + 8);
        v16bf a = __builtin_shufflevector(lo, hi, 0,1,2,3,4,5,6,7,8,9,10,11,12,13,14,15);
        st[t] = __builtin_amdgcn_wmma_f32_16x16x32_bf16(
            false, a, false, qB[hc], (short)0, st[t], false, false);
      }
    }
    // ---- hoist V^T tile loads so global latency overlaps softmax VALU ------
    v8bf vlo[4], vhi[4];
#pragma unroll
    for (int ht = 0; ht < 4; ++ht) {
      const __bf16* vrow = VT + (size_t)(ht * 16 + n) * 2048 + kb;
      vlo[ht] = *(const v8bf*)(vrow + 16 * half);
      vhi[ht] = *(const v8bf*)(vrow + 16 * half + 8);
    }
    // ---- causal mask + online softmax (lane holds one query column) -------
    float cmax = -__builtin_inff();
#pragma unroll
    for (int t = 0; t < 2; ++t)
#pragma unroll
      for (int r = 0; r < 8; ++r) {
        int kk = kb + t * 16 + 8 * half + r;
        float v = (kk > qq) ? -__builtin_inff() : st[t][r];
        st[t][r] = v;
        cmax = fmaxf(cmax, v);
      }
    cmax = fmaxf(cmax, __shfl_xor(cmax, 16, 32));
    float mnew  = fmaxf(m, cmax);
    float alpha = __builtin_exp2f((m - mnew) * SC_LOG2E);  // 0 on first chunk
    float psum = 0.f;
#pragma unroll
    for (int t = 0; t < 2; ++t)
#pragma unroll
      for (int r = 0; r < 8; ++r) {
        float p = __builtin_exp2f((st[t][r] - mnew) * SC_LOG2E);
        st[t][r] = p;
        psum += p;
      }
    psum += __shfl_xor(psum, 16, 32);
    lsum = lsum * alpha + psum;
    m = mnew;
#pragma unroll
    for (int j = 0; j < 4; ++j)
#pragma unroll
      for (int r = 0; r < 8; ++r) o[j][r] *= alpha;

    // ---- stage P^T to LDS as [q][k] bf16 (same-wave DS ops stay in order) --
    {
      uint32_t* dst = (uint32_t*)(lds + n * 32);
#pragma unroll
      for (int t = 0; t < 2; ++t) {
        int kloc = t * 16 + 8 * half;
#pragma unroll
        for (int r = 0; r < 8; r += 2)
          dst[(kloc + r) >> 1] = cvt_pk2(st[t][r], st[t][r + 1]);
      }
    }
    asm volatile("" ::: "memory");  // keep DS load after DS stores
    v16bf pB = *(const v16bf*)(lds + n * 32 + 16 * half);

    // ---- O^T += V^T_tile (16h x 32k) * P^T (32k x 16q) ---------------------
#pragma unroll
    for (int ht = 0; ht < 4; ++ht) {
      v16bf a = __builtin_shufflevector(vlo[ht], vhi[ht],
                                        0,1,2,3,4,5,6,7,8,9,10,11,12,13,14,15);
      o[ht] = __builtin_amdgcn_wmma_f32_16x16x32_bf16(
          false, a, false, pB, (short)0, o[ht], false, false);
    }
  }

  // ---- epilogue: out[b][q][h] = O^T[h][q] / lsum ---------------------------
  float inv = 1.0f / lsum;
  float* orow = Out + ((size_t)b * 2048 + qq) * 64;
#pragma unroll
  for (int ht = 0; ht < 4; ++ht) {
    int hbase = ht * 16 + 8 * half;
    v4f a = { o[ht][0] * inv, o[ht][1] * inv, o[ht][2] * inv, o[ht][3] * inv };
    v4f c = { o[ht][4] * inv, o[ht][5] * inv, o[ht][6] * inv, o[ht][7] * inv };
    *(v4f*)(orow + hbase)     = a;
    *(v4f*)(orow + hbase + 4) = c;
  }
}

// ---------------- Launch ------------------------------------------------------
extern "C" void kernel_launch(void* const* d_in, const int* in_sizes, int n_in,
                              void* d_out, int out_size, void* d_ws, size_t ws_size,
                              hipStream_t stream) {
  const float* Xk = (const float*)d_in[0];
  const float* Xv = (const float*)d_in[1];
  const float* Xq = (const float*)d_in[2];
  const float* Wk = (const float*)d_in[3];
  const float* Wv = (const float*)d_in[4];
  const float* Wq = (const float*)d_in[5];

  uint8_t* ws = (uint8_t*)d_ws;
  __bf16* Wtq = (__bf16*)(ws);                       // 64*1024*2 = 128 KiB each
  __bf16* Wtk = (__bf16*)(ws + 131072);
  __bf16* Wtv = (__bf16*)(ws + 262144);
  __bf16* Qb  = (__bf16*)(ws + 393216);              // [8*2048][64] bf16 = 2 MiB
  __bf16* Kb  = (__bf16*)(ws + 393216 + 2097152);
  __bf16* Vt  = (__bf16*)(ws + 393216 + 2 * 2097152); // [8][64][2048] bf16

  wt_prep<<<256, 256, 0, stream>>>(Wq, Wtq);
  wt_prep<<<256, 256, 0, stream>>>(Wk, Wtk);
  wt_prep<<<256, 256, 0, stream>>>(Wv, Wtv);

  proj_kernel<<<128, 256, 0, stream>>>(Xq, Wtq, Qb, nullptr);
  proj_kernel<<<128, 256, 0, stream>>>(Xk, Wtk, Kb, nullptr);
  proj_kernel<<<128, 256, 0, stream>>>(Xv, Wtv, nullptr, Vt);

  attn_kernel<<<dim3(16, 8), 256, 0, stream>>>(Qb, Kb, Vt, (float*)d_out);
}
